// Multi_Self_Att_54219667145436
// MI455X (gfx1250) — matrix operand order repaired and checked by reading, not compiled, not run on hardware
//
#include <hip/hip_runtime.h>
#include <hip/hip_bf16.h>
#include <math.h>

typedef __attribute__((ext_vector_type(16))) _Float16 v16h;
typedef __attribute__((ext_vector_type(8)))  _Float16 v8h;
typedef __attribute__((ext_vector_type(4)))  _Float16 v4h;
typedef __attribute__((ext_vector_type(8)))  float    v8f;
typedef __attribute__((ext_vector_type(4)))  float    v4f;

#define Bq  4
#define Lq  2048
#define Dq  512
#define Hq  8
#define DKq 64
#define Mq  (Bq*Lq)   // 8192 rows
#define LPAD 40       // padded LDS row stride in halfs (80B: 16B-aligned, bank-friendly)

union AB { v16h v; v8h h[2]; };

__device__ __forceinline__ v8f wmma_f16(v16h a, v16h b, v8f c) {
  // D = A(16x32 f16) * B(32x16 f16) + C(16x16 f32)
  return __builtin_amdgcn_wmma_f32_16x16x32_f16(false, a, false, b, (short)0, c,
                                                false, false);
}

// Async DMA: global -> LDS, 16 bytes per lane, tracked by ASYNCcnt.
__device__ __forceinline__ void async_b128(_Float16* lds_dst, const _Float16* gsrc) {
  unsigned loff = (unsigned)(unsigned long long)lds_dst;  // low 32 bits = LDS offset
  asm volatile("global_load_async_to_lds_b128 %0, %1, off"
               :: "v"(loff), "v"(gsrc) : "memory");
}
__device__ __forceinline__ void wait_async0() {
  asm volatile("s_wait_asynccnt 0x0" ::: "memory");
}

// ---------------------------------------------------------------------------
// fp32 -> fp16 conversion (vectorized x4)
// ---------------------------------------------------------------------------
__global__ __launch_bounds__(256)
void cvt_f32_f16(const v4f* __restrict__ in, v4h* __restrict__ out, int n4) {
  int i = blockIdx.x * 256 + threadIdx.x;
  if (i < n4) out[i] = __builtin_convertvector(in[i], v4h);
}

// ---------------------------------------------------------------------------
// WMMA GEMM with async-LDS double-buffered staging:
//   C[M x 512] = A[M x 512] @ W^T + bias, W row-major along K ([n][k]).
// Block: 256 thr = 8 waves -> 32(M) x 256(N); wave computes 16 x 64.
// Per K-step (32), the block stages A-tile 32x32 and W-tile 256x32 into LDS
// via global_load_async_to_lds_b128 (5 instr/thread, wave-uniform), then all
// fragments come from LDS. Transfer(k+1) overlaps compute(k).
// MODE 0: f16 row-major out     MODE 1: f16 scatter [B,H,L,DK] (q/k)
// MODE 2: f16 scatter [B,H,DK,L] (v transposed)   MODE 3: f32 row-major out
// ---------------------------------------------------------------------------
template<int MODE>
__global__ __launch_bounds__(256)
void gemm_wmma(const _Float16* __restrict__ A, const _Float16* __restrict__ Wm,
               const float* __restrict__ bias, _Float16* __restrict__ outH,
               float* __restrict__ outF, float scale)
{
  const int tid  = threadIdx.x;
  const int lane = tid & 31;
  const int wave = tid >> 5;
  const int lh = lane >> 4, ll = lane & 15;
  const int mw = wave & 1, nw = wave >> 1;
  const int m0 = blockIdx.x * 32 + mw * 16;
  const int n0base = blockIdx.y * 256;
  const int n0 = n0base + nw * 64;

  __shared__ __align__(16) _Float16 ldsW[2][256 * LPAD];  // 2 x 20KB
  __shared__ __align__(16) _Float16 ldsA[2][32 * LPAD];   // 2 x 2.5KB

  // stage one 32-wide K-slice of A-tile + W-tile into LDS buffer `buf`
  auto issue = [&](int buf, int kk) {
    #pragma unroll
    for (int c = 0; c < 4; ++c) {                 // 1024 W chunks of 16B
      const int w = tid + 256 * c;
      const int n = w >> 2, col = w & 3;
      async_b128(&ldsW[buf][n * LPAD + col * 8],
                 Wm + (size_t)(n0base + n) * Dq + kk + col * 8);
    }
    {                                             // 128 A chunks (duplicated x2)
      const int a = tid & 127;
      const int ar = a >> 2, ac = a & 3;
      async_b128(&ldsA[buf][ar * LPAD + ac * 8],
                 A + (size_t)(blockIdx.x * 32 + ar) * Dq + kk + ac * 8);
    }
  };

  v8f acc[4] = {};
  issue(0, 0);

  for (int step = 0; step < 16; ++step) {
    const int buf = step & 1;
    wait_async0();        // this wave's part of batch `step` is in LDS
    __syncthreads();      // all waves' parts are in LDS; prior readers done
    if (step < 15) issue(buf ^ 1, (step + 1) * 32);

    // A frag: lanes 0-15 row=ll K{0..15}; lanes 16-31 K{16..31}
    AB a;
    const _Float16* arow = &ldsA[buf][(mw * 16 + ll) * LPAD];
    a.h[0] = *(const v8h*)(arow + lh * 16);
    a.h[1] = *(const v8h*)(arow + lh * 16 + 8);
    #pragma unroll
    for (int t = 0; t < 4; ++t) {
      // B frag: lane col N, K split 0..15 / 16..31 across lane halves
      const _Float16* wrow = &ldsW[buf][(nw * 64 + t * 16 + ll) * LPAD + lh * 16];
      AB b;
      b.h[0] = *(const v8h*)(wrow);
      b.h[1] = *(const v8h*)(wrow + 8);
      acc[t] = wmma_f16(a.v, b.v, acc[t]);
    }
  }

  #pragma unroll
  for (int t = 0; t < 4; ++t) {
    const int n = n0 + t * 16 + ll;
    const float bv = bias[n];
    #pragma unroll
    for (int r = 0; r < 8; ++r) {
      const int m = m0 + r + lh * 8;           // C layout: VGPR r, lane half
      const float v = (acc[t][r] + bv) * scale;
      if (MODE == 0) {
        outH[(size_t)m * Dq + n] = (_Float16)v;
      } else if (MODE == 1) {
        const int b = m >> 11, l = m & (Lq - 1), h = n >> 6, dk = n & 63;
        outH[(((size_t)(b * Hq + h)) * Lq + l) * DKq + dk] = (_Float16)v;
      } else if (MODE == 2) {
        const int b = m >> 11, l = m & (Lq - 1), h = n >> 6, dk = n & 63;
        outH[(((size_t)(b * Hq + h)) * DKq + dk) * Lq + l] = (_Float16)v;
      } else {
        outF[(size_t)m * Dq + n] = v;
      }
    }
  }
}

// ---------------------------------------------------------------------------
// Flash-style masked attention. One wave owns 16 query rows, block = 8 waves
// = 128 rows of one (b,h). Online softmax in f32, P/V matmul via WMMA with
// LDS transpose of the probability tile (C-layout -> A-layout).
// q16/k16: [B,H,L,DK] f16 (q pre-scaled by 1/sqrt(DK)); vT16: [B,H,DK,L] f16.
// ---------------------------------------------------------------------------
__global__ __launch_bounds__(256)
void attn_fwd(const _Float16* __restrict__ q16, const _Float16* __restrict__ k16,
              const _Float16* __restrict__ vT16, const int* __restrict__ mask,
              _Float16* __restrict__ attOut)
{
  const int lane = threadIdx.x & 31, wave = threadIdx.x >> 5;
  const int lh = lane >> 4, ll = lane & 15;
  const int bh = blockIdx.y;
  const int b = bh >> 3, h = bh & 7;
  const int row0 = blockIdx.x * 128 + wave * 16;

  __shared__ __align__(32) _Float16 pbuf[8][16][32];   // per-wave 16x32 P tile

  const _Float16* qrow = q16 + ((size_t)bh * Lq + row0 + ll) * DKq;
  AB qa0, qa1;
  qa0.h[0] = *(const v8h*)(qrow +      lh * 16);
  qa0.h[1] = *(const v8h*)(qrow + 8  + lh * 16);
  qa1.h[0] = *(const v8h*)(qrow + 32 + lh * 16);
  qa1.h[1] = *(const v8h*)(qrow + 40 + lh * 16);

  v8f acc[4] = {};
  float rowm[8], rowl[8];
  #pragma unroll
  for (int r = 0; r < 8; ++r) { rowm[r] = -1.0e30f; rowl[r] = 0.f; }

  const int* maskB = mask + (size_t)b * Lq * Lq;

  for (int j0 = 0; j0 < Lq; j0 += 32) {
    if (j0 + 32 < Lq) {   // global_prefetch for the next chunk
      __builtin_prefetch(k16 + ((size_t)bh * Lq + j0 + 32 + ll) * DKq, 0, 3);
      __builtin_prefetch(vT16 + ((size_t)bh * DKq + ll) * Lq + j0 + 32, 0, 3);
      __builtin_prefetch(maskB + (size_t)(row0 + ll) * Lq + j0 + 32, 0, 0);
    }
    // ---- scores: 16x32 tile = two 16x16 column tiles, 2 WMMAs each ----
    v8f S[2] = {};
    #pragma unroll
    for (int t = 0; t < 2; ++t) {
      const _Float16* krow = k16 + ((size_t)bh * Lq + j0 + t * 16 + ll) * DKq;
      v16h kb0 = *(const v16h*)(krow +      lh * 16);
      v16h kb1 = *(const v16h*)(krow + 32 + lh * 16);
      S[t] = wmma_f16(qa0.v, kb0, S[t]);
      S[t] = wmma_f16(qa1.v, kb1, S[t]);
    }
    // ---- mask + online softmax (rows split across lane halves) ----
    #pragma unroll
    for (int r = 0; r < 8; ++r) {
      const int m = row0 + r + 8 * lh;
      const int* mr = maskB + (size_t)m * Lq + j0;
      const int f0 = mr[ll], f1 = mr[16 + ll];
      float s0 = f0 ? S[0][r] : -1.0e9f;
      float s1 = f1 ? S[1][r] : -1.0e9f;
      float mx = fmaxf(s0, s1);
      #pragma unroll
      for (int o = 8; o > 0; o >>= 1) mx = fmaxf(mx, __shfl_xor(mx, o, 16));
      const float newm = fmaxf(rowm[r], mx);
      const float resc = __expf(rowm[r] - newm);
      rowm[r] = newm;
      const float e0 = __expf(s0 - newm);
      const float e1 = __expf(s1 - newm);
      float rsum = e0 + e1;                   // denominator keeps masked terms
      #pragma unroll
      for (int o = 8; o > 0; o >>= 1) rsum += __shfl_xor(rsum, o, 16);
      rowl[r] = rowl[r] * resc + rsum;
      #pragma unroll
      for (int t = 0; t < 4; ++t) acc[t][r] *= resc;
      pbuf[wave][r + 8 * lh][ll]      = (_Float16)(f0 ? e0 : 0.f);  // att*mask
      pbuf[wave][r + 8 * lh][16 + ll] = (_Float16)(f1 ? e1 : 0.f);
    }
    asm volatile("s_wait_dscnt 0x0" ::: "memory");   // same-wave LDS RAW
    // ---- reload P as A-fragment (transpose C-layout -> A-layout) ----
    AB pa;
    pa.h[0] = *(const v8h*)&pbuf[wave][ll][lh * 16];
    pa.h[1] = *(const v8h*)&pbuf[wave][ll][lh * 16 + 8];
    // ---- P @ V : v stored transposed, so B-frags are contiguous ----
    #pragma unroll
    for (int t = 0; t < 4; ++t) {
      const _Float16* vrow =
          vT16 + ((size_t)bh * DKq + t * 16 + ll) * Lq + j0 + lh * 16;
      v16h vb = *(const v16h*)vrow;
      acc[t] = wmma_f16(pa.v, vb, acc[t]);
    }
  }

  #pragma unroll
  for (int r = 0; r < 8; ++r) {
    const float inv = 1.f / rowl[r];
    const int m = row0 + r + 8 * lh;
    #pragma unroll
    for (int t = 0; t < 4; ++t) {
      attOut[((size_t)b * Lq + m) * Dq + h * DKq + t * 16 + ll] =
          (_Float16)(acc[t][r] * inv);
    }
  }
}

// ---------------------------------------------------------------------------
// LayerNorm helpers (ddof=1 variance, y = g*(x-mean)/(sd+eps)+b)
// ---------------------------------------------------------------------------
__device__ __forceinline__ void block_reduce2(float& s, float& q,
                                              float* rs, float* rq)
{
  #pragma unroll
  for (int o = 16; o > 0; o >>= 1) {
    s += __shfl_xor(s, o, 32);
    q += __shfl_xor(q, o, 32);
  }
  const int lane = threadIdx.x & 31, w = threadIdx.x >> 5;
  if (lane == 0) { rs[w] = s; rq[w] = q; }
  __syncthreads();
  float ts = 0.f, tq = 0.f;
  #pragma unroll
  for (int i = 0; i < 8; ++i) { ts += rs[i]; tq += rq[i]; }
  s = ts; q = tq;
  __syncthreads();
}

// feed1 = x + LN(proj_raw); write f32 (for later residual) and f16 (for GEMM)
__global__ __launch_bounds__(256)
void ln_residual(const float* __restrict__ x, const float* __restrict__ raw,
                 const float* __restrict__ gamma, const float* __restrict__ beta,
                 float* __restrict__ f1f, _Float16* __restrict__ f1h)
{
  __shared__ float rs[8], rq[8];
  const int row = blockIdx.x, tid = threadIdx.x;
  const float* r = raw + (size_t)row * Dq;
  float a0 = r[tid], a1 = r[tid + 256];
  float s = a0 + a1, q = a0 * a0 + a1 * a1;
  block_reduce2(s, q, rs, rq);
  const float mean = s * (1.f / 512.f);
  const float sd = sqrtf(fmaxf((q - 512.f * mean * mean) * (1.f / 511.f), 0.f));
  const float inv = 1.f / (sd + 1e-6f);
  const float* xr = x + (size_t)row * Dq;
  const float y0 = xr[tid]       + gamma[tid]       * (a0 - mean) * inv + beta[tid];
  const float y1 = xr[tid + 256] + gamma[tid + 256] * (a1 - mean) * inv + beta[tid + 256];
  f1f[(size_t)row * Dq + tid]       = y0;
  f1f[(size_t)row * Dq + tid + 256] = y1;
  f1h[(size_t)row * Dq + tid]       = (_Float16)y0;
  f1h[(size_t)row * Dq + tid + 256] = (_Float16)y1;
}

// inner = LN(g4raw); feed2 = LN(feed1 + inner)
__global__ __launch_bounds__(256)
void ln_ffn(const float* __restrict__ g4raw, const float* __restrict__ f1f,
            const float* __restrict__ gamma, const float* __restrict__ beta,
            float* __restrict__ out)
{
  __shared__ float rs[8], rq[8];
  const int row = blockIdx.x, tid = threadIdx.x;
  const float* g = g4raw + (size_t)row * Dq;
  float a0 = g[tid], a1 = g[tid + 256];
  float s = a0 + a1, q = a0 * a0 + a1 * a1;
  block_reduce2(s, q, rs, rq);
  float mean = s * (1.f / 512.f);
  float sd = sqrtf(fmaxf((q - 512.f * mean * mean) * (1.f / 511.f), 0.f));
  float inv = 1.f / (sd + 1e-6f);
  const float i0 = gamma[tid]       * (a0 - mean) * inv + beta[tid];
  const float i1 = gamma[tid + 256] * (a1 - mean) * inv + beta[tid + 256];
  const float* f = f1f + (size_t)row * Dq;
  const float t0 = f[tid] + i0, t1 = f[tid + 256] + i1;
  s = t0 + t1; q = t0 * t0 + t1 * t1;
  block_reduce2(s, q, rs, rq);
  mean = s * (1.f / 512.f);
  sd = sqrtf(fmaxf((q - 512.f * mean * mean) * (1.f / 511.f), 0.f));
  inv = 1.f / (sd + 1e-6f);
  out[(size_t)row * Dq + tid]       = gamma[tid]       * (t0 - mean) * inv + beta[tid];
  out[(size_t)row * Dq + tid + 256] = gamma[tid + 256] * (t1 - mean) * inv + beta[tid + 256];
}

// ---------------------------------------------------------------------------
extern "C" void kernel_launch(void* const* d_in, const int* in_sizes, int n_in,
                              void* d_out, int out_size, void* d_ws, size_t ws_size,
                              hipStream_t stream)
{
  const float* x     = (const float*)d_in[0];
  const int*   mask  = (const int*)  d_in[1];
  const float* W     = (const float*)d_in[2];
  const float* bL    = (const float*)d_in[3];
  const float* gamma = (const float*)d_in[4];
  const float* beta  = (const float*)d_in[5];
  float* out = (float*)d_out;
  (void)in_sizes; (void)n_in; (void)out_size; (void)ws_size;

  char* p = (char*)d_ws;
  size_t off = 0;
  auto take = [&](size_t bytes) -> char* {
    char* r = p + off;
    off += (bytes + 255) & ~(size_t)255;
    return r;
  };

  _Float16* W16  = (_Float16*)take((size_t)6 * Dq * Dq * 2);
  _Float16* x16  = (_Float16*)take((size_t)Mq * Dq * 2);
  _Float16* q16  = (_Float16*)take((size_t)Mq * Dq * 2);
  _Float16* k16  = (_Float16*)take((size_t)Mq * Dq * 2);
  _Float16* vT16 = (_Float16*)take((size_t)Mq * Dq * 2);
  _Float16* ao16 = (_Float16*)take((size_t)Mq * Dq * 2);
  float*    praw = (float*)   take((size_t)Mq * Dq * 4);
  float*    f1f  = (float*)   take((size_t)Mq * Dq * 4);
  _Float16* f1h  = (_Float16*)take((size_t)Mq * Dq * 2);
  _Float16* t5   = (_Float16*)take((size_t)Mq * Dq * 2);
  float*    g4   = (float*)   take((size_t)Mq * Dq * 4);

  dim3 blk(256);
  { int n4 = (6 * Dq * Dq) / 4;
    cvt_f32_f16<<<(n4 + 255) / 256, blk, 0, stream>>>((const v4f*)W, (v4h*)W16, n4); }
  { int n4 = (Mq * Dq) / 4;
    cvt_f32_f16<<<(n4 + 255) / 256, blk, 0, stream>>>((const v4f*)x, (v4h*)x16, n4); }

  dim3 gG(Mq / 32, Dq / 256);
  // q,k,v projections (q pre-scaled by 1/sqrt(DK)=0.125; v stored transposed)
  gemm_wmma<1><<<gG, blk, 0, stream>>>(x16, W16 + 0 * Dq * Dq, bL + 0 * Dq, q16,  nullptr, 0.125f);
  gemm_wmma<1><<<gG, blk, 0, stream>>>(x16, W16 + 1 * Dq * Dq, bL + 1 * Dq, k16,  nullptr, 1.f);
  gemm_wmma<2><<<gG, blk, 0, stream>>>(x16, W16 + 2 * Dq * Dq, bL + 2 * Dq, vT16, nullptr, 1.f);
  // masked flash attention
  attn_fwd<<<dim3(Lq / 128, Bq * Hq), blk, 0, stream>>>(q16, k16, vT16, mask, ao16);
  // output projection (raw f32 for LN)
  gemm_wmma<3><<<gG, blk, 0, stream>>>(ao16, W16 + 3 * Dq * Dq, bL + 3 * Dq, nullptr, praw, 1.f);
  // feed1 = x + LN(proj)
  ln_residual<<<Mq, blk, 0, stream>>>(x, praw, gamma, beta, f1f, f1h);
  // FFN: lin5 then lin4
  gemm_wmma<0><<<gG, blk, 0, stream>>>(f1h, W16 + 5 * Dq * Dq, bL + 5 * Dq, t5, nullptr, 1.f);
  gemm_wmma<3><<<gG, blk, 0, stream>>>(t5,  W16 + 4 * Dq * Dq, bL + 4 * Dq, nullptr, g4, 1.f);
  // feed2 = LN(feed1 + LN(g4))
  ln_ffn<<<Mq, blk, 0, stream>>>(g4, f1f, gamma, beta, out);
}
